// SwinTransformerBlock_14027363188955
// MI455X (gfx1250) — matrix operand order repaired and checked by reading, not compiled, not run on hardware
//
#include <hip/hip_runtime.h>
#include <hip/hip_bf16.h>

typedef __attribute__((ext_vector_type(16))) _Float16 v16h;
typedef __attribute__((ext_vector_type(8)))  _Float16 v8h;
typedef __attribute__((ext_vector_type(8)))  float    v8f;
typedef __attribute__((ext_vector_type(4)))  int      v4i;

// Problem constants
#define BATCH 32
#define HH 56
#define WW 56
#define CC 384
#define HEADS 12
#define DH 32
#define WIN 7
#define SHIFT 3
#define NTOK 49
#define NWIN 64          // windows per image (8x8)
#define TOKENS 100352    // BATCH * HH * WW
#define HID 1536
#define QKV_N 1152
#define ATT_SCALE 0.17677669529663689f  // 32^-0.5

#define WMMA16(a, b, c) \
  __builtin_amdgcn_wmma_f32_16x16x32_f16(false, (a), false, (b), (short)0, (c), false, false)

// ---- async global->LDS path (gfx1250), with synchronous fallback -------------
#if defined(__AMDGCN__) && __has_builtin(__builtin_amdgcn_global_load_async_to_lds_b128)
#define HAVE_ASYNC_LDS 1
typedef __attribute__((address_space(1))) v4i as1_v4i;   // global int4
typedef __attribute__((address_space(3))) v4i as3_v4i;   // LDS int4
#define GPTR(p) ((as1_v4i*)(p))
#define LPTR(p) ((as3_v4i*)(p))
#else
#define HAVE_ASYNC_LDS 0
#endif

__device__ __forceinline__ void async_wait_all() {
#if HAVE_ASYNC_LDS
#if __has_builtin(__builtin_amdgcn_s_wait_asynccnt)
  __builtin_amdgcn_s_wait_asynccnt(0);
#else
  asm volatile("s_wait_asynccnt 0x0" ::: "memory");
#endif
#endif
}

// ---------------- fragment loaders (16-bit A/B layouts, wave32) ----------------
// A 16x32 (MxK): lane l -> row = row0 + (l&15); halves A[row][k0..k0+7], A[row][k0+16..k0+23], k0 = 8*(l>>4)
__device__ __forceinline__ v16h frag_a(const _Float16* mat, int ld, int row0, int kbase) {
  int lane = threadIdx.x & 31;
  int r = row0 + (lane & 15);
  int k0 = kbase + ((lane >> 4) << 3);
  const _Float16* p = mat + (size_t)r * ld + k0;
  v8h lo = *(const v8h*)(p);
  v8h hi = *(const v8h*)(p + 16);
  return __builtin_shufflevector(lo, hi, 0,1,2,3,4,5,6,7,8,9,10,11,12,13,14,15);
}
__device__ __forceinline__ v16h frag_a_cl(const _Float16* mat, int ld, int row0, int kbase, int rmax) {
  int lane = threadIdx.x & 31;
  int r = row0 + (lane & 15); if (r > rmax) r = rmax;
  int k0 = kbase + ((lane >> 4) << 3);
  const _Float16* p = mat + (size_t)r * ld + k0;
  v8h lo = *(const v8h*)(p);
  v8h hi = *(const v8h*)(p + 16);
  return __builtin_shufflevector(lo, hi, 0,1,2,3,4,5,6,7,8,9,10,11,12,13,14,15);
}
// B 32x16 (KxN), source stored K-contiguous as [N][K]: lane l -> col = col0 + (l&15);
// halves B[col][k0..k0+7], B[col][k0+16..k0+23], k0 = 8*(l>>4)
__device__ __forceinline__ v16h frag_b(const _Float16* mat, int ld, int col0, int kbase) {
  int lane = threadIdx.x & 31;
  int c = col0 + (lane & 15);
  int k0 = kbase + ((lane >> 4) << 3);
  const _Float16* p = mat + (size_t)c * ld + k0;
  v8h lo = *(const v8h*)(p);
  v8h hi = *(const v8h*)(p + 16);
  return __builtin_shufflevector(lo, hi, 0,1,2,3,4,5,6,7,8,9,10,11,12,13,14,15);
}
__device__ __forceinline__ v16h frag_b_cl(const _Float16* mat, int ld, int col0, int kbase, int cmax) {
  int lane = threadIdx.x & 31;
  int c = col0 + (lane & 15); if (c > cmax) c = cmax;
  int k0 = kbase + ((lane >> 4) << 3);
  const _Float16* p = mat + (size_t)c * ld + k0;
  v8h lo = *(const v8h*)(p);
  v8h hi = *(const v8h*)(p + 16);
  return __builtin_shufflevector(lo, hi, 0,1,2,3,4,5,6,7,8,9,10,11,12,13,14,15);
}

// ---------------- weight convert+transpose: f32 [K][N] -> f16 [N][K] ----------------
__global__ void cvt_w_kernel(const float* __restrict__ src, _Float16* __restrict__ dst, int K, int N) {
  int i = blockIdx.x * 256 + threadIdx.x;
  if (i < K * N) {
    int k = i / N, n = i % N;
    dst[(size_t)n * K + k] = (_Float16)src[i];
  }
}

// ---------------- LayerNorm (+optional shift+window partition), f32 -> f16 ----------------
template<bool SHIFT_WIN>
__global__ __launch_bounds__(128) void ln_kernel(const float* __restrict__ in,
                                                 const float* __restrict__ gw,
                                                 const float* __restrict__ gb,
                                                 _Float16* __restrict__ out) {
  __shared__ float red[128];
  int row = blockIdx.x, tid = threadIdx.x;
  size_t sp;
  if (SHIFT_WIN) {
    int win = row / NTOK, n = row % NTOK;
    int b = win >> 6, wrc = win & 63, wr = wrc >> 3, wc = wrc & 7;
    int iy = n / WIN, ix = n % WIN;
    int y = (wr * WIN + iy + SHIFT) % HH;
    int x = (wc * WIN + ix + SHIFT) % WW;
    sp = ((size_t)b * HH + y) * WW + x;
  } else {
    sp = (size_t)row;
  }
  const float* src = in + sp * CC;
  float v0 = src[tid], v1 = src[tid + 128], v2 = src[tid + 256];
  red[tid] = v0 + v1 + v2;
  __syncthreads();
  for (int off = 64; off; off >>= 1) { if (tid < off) red[tid] += red[tid + off]; __syncthreads(); }
  float mean = red[0] * (1.0f / CC);
  __syncthreads();
  float d0 = v0 - mean, d1 = v1 - mean, d2 = v2 - mean;
  red[tid] = d0 * d0 + d1 * d1 + d2 * d2;
  __syncthreads();
  for (int off = 64; off; off >>= 1) { if (tid < off) red[tid] += red[tid + off]; __syncthreads(); }
  float rstd = rsqrtf(red[0] * (1.0f / CC) + 1e-5f);
  _Float16* dst = out + (size_t)row * CC;
  dst[tid]       = (_Float16)(d0 * rstd * gw[tid]       + gb[tid]);
  dst[tid + 128] = (_Float16)(d1 * rstd * gw[tid + 128] + gb[tid + 128]);
  dst[tid + 256] = (_Float16)(d2 * rstd * gw[tid + 256] + gb[tid + 256]);
}

// ---------------- tiled WMMA GEMM: A[M][K] f16 x Bt[N][K] f16 -> epilogue ----------------
enum { EPI_QKV = 0, EPI_PROJ = 1, EPI_FC1 = 2, EPI_FC2 = 3 };

// stage one 128x32 tile pair into LDS (async DMA when available)
__device__ __forceinline__ void stage_tiles(const _Float16* __restrict__ A,
                                            const _Float16* __restrict__ Bt,
                                            int K, int m0, int n0, int k0,
                                            _Float16 (*as)[40], _Float16 (*bs)[40],
                                            int tid) {
#pragma unroll
  for (int p = 0; p < 2; ++p) {
    int cid = tid + p * 256;                  // 512 chunks of 8 halves each
    int row = cid >> 2, kc = (cid & 3) << 3;
#if HAVE_ASYNC_LDS
    __builtin_amdgcn_global_load_async_to_lds_b128(
        GPTR(A + (size_t)(m0 + row) * K + k0 + kc), LPTR(&as[row][kc]), 0, 0);
    __builtin_amdgcn_global_load_async_to_lds_b128(
        GPTR(Bt + (size_t)(n0 + row) * K + k0 + kc), LPTR(&bs[row][kc]), 0, 0);
#else
    *(v8h*)&as[row][kc] = *(const v8h*)(A  + (size_t)(m0 + row) * K + k0 + kc);
    *(v8h*)&bs[row][kc] = *(const v8h*)(Bt + (size_t)(n0 + row) * K + k0 + kc);
#endif
  }
}

template<int EPI>
__global__ __launch_bounds__(256) void gemm_wmma(const _Float16* __restrict__ A,
                                                 const _Float16* __restrict__ Bt,
                                                 const float* __restrict__ bias,
                                                 const float* __restrict__ extra,
                                                 void* __restrict__ outp,
                                                 int N, int K) {
  __shared__ _Float16 As[2][128][40];   // double-buffered 128x32 tiles (+pad, 16B-aligned rows)
  __shared__ _Float16 Bs[2][128][40];
  int tid = threadIdx.x, lane = tid & 31, wv = tid >> 5;
  int waveM = wv & 3, waveN = wv >> 2;        // 4x2 wave grid: 32 rows x 64 cols each
  int m0 = blockIdx.x * 128, n0 = blockIdx.y * 128;

  v8f zero = {};
  v8f acc[2][4];
#pragma unroll
  for (int mt = 0; mt < 2; ++mt)
#pragma unroll
    for (int nt = 0; nt < 4; ++nt) acc[mt][nt] = zero;

  int nk = K >> 5;
  stage_tiles(A, Bt, K, m0, n0, 0, As[0], Bs[0], tid);
  async_wait_all();
  __syncthreads();

  for (int kt = 0; kt < nk; ++kt) {
    int cur = kt & 1;
    if (kt + 1 < nk) {
      stage_tiles(A, Bt, K, m0, n0, (kt + 1) << 5, As[cur ^ 1], Bs[cur ^ 1], tid);
#if !HAVE_ASYNC_LDS
      __builtin_prefetch(A  + (size_t)(m0 + (tid >> 1)) * K + ((kt + 1) << 5), 0, 1);
      __builtin_prefetch(Bt + (size_t)(n0 + (tid >> 1)) * K + ((kt + 1) << 5), 0, 1);
#endif
    }
    v16h af[2], bf[4];
#pragma unroll
    for (int mt = 0; mt < 2; ++mt) af[mt] = frag_a(&As[cur][0][0], 40, waveM * 32 + mt * 16, 0);
#pragma unroll
    for (int nt = 0; nt < 4; ++nt) bf[nt] = frag_b(&Bs[cur][0][0], 40, waveN * 64 + nt * 16, 0);
#pragma unroll
    for (int mt = 0; mt < 2; ++mt)
#pragma unroll
      for (int nt = 0; nt < 4; ++nt) acc[mt][nt] = WMMA16(af[mt], bf[nt], acc[mt][nt]);
    async_wait_all();          // next tile landed in the alternate buffer
    __syncthreads();           // everyone done reading current buffer
  }

#pragma unroll
  for (int mt = 0; mt < 2; ++mt)
#pragma unroll
    for (int nt = 0; nt < 4; ++nt) {
      int col = n0 + waveN * 64 + nt * 16 + (lane & 15);
      float bcol = bias[col];
#pragma unroll
      for (int i = 0; i < 8; ++i) {
        int row = m0 + waveM * 32 + mt * 16 + i + ((lane >> 4) << 3);
        float v = acc[mt][nt][i] + bcol;
        if constexpr (EPI == EPI_QKV) {
          ((_Float16*)outp)[(size_t)row * N + col] = (_Float16)v;
        } else if constexpr (EPI == EPI_FC1) {
          float g = 0.5f * v * (1.0f + erff(v * 0.70710678f));
          ((_Float16*)outp)[(size_t)row * N + col] = (_Float16)g;
        } else if constexpr (EPI == EPI_FC2) {
          ((float*)outp)[(size_t)row * N + col] = v + extra[(size_t)row * N + col];
        } else { // EPI_PROJ: window reverse + un-shift + residual add into x1
          int win = row / NTOK, n = row % NTOK;
          int b = win >> 6, wrc = win & 63, wr = wrc >> 3, wc = wrc & 7;
          int iy = n / WIN, ix = n % WIN;
          int y = (wr * WIN + iy + SHIFT) % HH;
          int x = (wc * WIN + ix + SHIFT) % WW;
          size_t sp = ((size_t)b * HH + y) * WW + x;
          ((float*)outp)[sp * CC + col] = v + extra[sp * CC + col];
        }
      }
    }
}

// ---------------- windowed attention: one wave per (window, head) ----------------
__global__ __launch_bounds__(64) void attn_kernel(const _Float16* __restrict__ qkv,
                                                  const float* __restrict__ rpb,
                                                  _Float16* __restrict__ out) {
  __shared__ _Float16 Vt[2][32][64];   // V transposed [d][n], cols >=49 zeroed
  __shared__ float    Sm[2][64][64];   // scores f32
  __shared__ _Float16 Pm[2][64][64];   // softmax probs f16, padded with zeros
  int wv = threadIdx.x >> 5, lane = threadIdx.x & 31;
  int w = blockIdx.x;
  int h = blockIdx.y * 2 + wv;

  const _Float16* qb = qkv + (size_t)w * NTOK * QKV_N + h * DH;         // Q
  const _Float16* kb = qb + CC;                                          // K
  const _Float16* vb = qb + 2 * CC;                                      // V

  for (int idx = lane; idx < 32 * 64; idx += 32) Vt[wv][idx >> 6][idx & 63] = (_Float16)0.0f;
  for (int idx = lane; idx < 64 * 64; idx += 32) {
    int rr = idx >> 6, cc = idx & 63;
    if (rr >= NTOK || cc >= NTOK) Pm[wv][rr][cc] = (_Float16)0.0f;
  }
  __syncthreads();
  for (int idx = lane; idx < NTOK * DH; idx += 32) {
    int n = idx >> 5, d = idx & 31;
    Vt[wv][d][n] = vb[(size_t)n * QKV_N + d];
  }
  __syncthreads();

  // S = Q K^T  (16x16 tiles, K=32 -> one wmma each; rows/cols clamped to 48)
#pragma unroll
  for (int mt = 0; mt < 4; ++mt) {
    v16h a = frag_a_cl(qb, QKV_N, mt * 16, 0, NTOK - 1);
#pragma unroll
    for (int nt = 0; nt < 4; ++nt) {
      v16h b = frag_b_cl(kb, QKV_N, nt * 16, 0, NTOK - 1);
      v8f c = {};
      c = WMMA16(a, b, c);
      int col = nt * 16 + (lane & 15);
#pragma unroll
      for (int i = 0; i < 8; ++i) Sm[wv][mt * 16 + i + ((lane >> 4) << 3)][col] = c[i];
    }
  }
  __syncthreads();

  // softmax with scale + relative position bias + shifted-window mask
  {
    int wrc = w & 63, wr = wrc >> 3, wc = wrc & 7;
    auto region = [&](int n) {
      int hy = wr * WIN + n / WIN;
      int wx = wc * WIN + n % WIN;
      int rr = hy < (HH - WIN) ? 0 : (hy < (HH - SHIFT) ? 1 : 2);
      int rc = wx < (WW - WIN) ? 0 : (wx < (WW - SHIFT) ? 1 : 2);
      return rr * 3 + rc;
    };
#pragma unroll
    for (int rp = 0; rp < 2; ++rp) {
      int r = lane + rp * 32;
      if (r < NTOK) {
        int regr = region(r);
        int qy = r / WIN, qx = r % WIN;
        float mx = -1e30f;
        for (int m = 0; m < NTOK; ++m) {
          int ky = m / WIN, kx = m % WIN;
          int ridx = (qy - ky + WIN - 1) * (2 * WIN - 1) + (qx - kx + WIN - 1);
          float s = Sm[wv][r][m] * ATT_SCALE + rpb[ridx * HEADS + h]
                  + ((region(m) != regr) ? -100.0f : 0.0f);
          Sm[wv][r][m] = s;
          mx = fmaxf(mx, s);
        }
        float sum = 0.0f;
        for (int m = 0; m < NTOK; ++m) {
          float e = __expf(Sm[wv][r][m] - mx);
          Sm[wv][r][m] = e;
          sum += e;
        }
        float inv = 1.0f / sum;
        for (int m = 0; m < NTOK; ++m) Pm[wv][r][m] = (_Float16)(Sm[wv][r][m] * inv);
      }
    }
  }
  __syncthreads();

  // O = P @ V  (4x2 tiles, K = 64 in two wmma steps)
#pragma unroll
  for (int mt = 0; mt < 4; ++mt) {
    v16h a0 = frag_a(&Pm[wv][0][0], 64, mt * 16, 0);
    v16h a1 = frag_a(&Pm[wv][0][0], 64, mt * 16, 32);
#pragma unroll
    for (int nt = 0; nt < 2; ++nt) {
      v8f c = {};
      c = WMMA16(a0, frag_b(&Vt[wv][0][0], 64, nt * 16, 0),  c);
      c = WMMA16(a1, frag_b(&Vt[wv][0][0], 64, nt * 16, 32), c);
      int col = nt * 16 + (lane & 15);
#pragma unroll
      for (int i = 0; i < 8; ++i) {
        int q = mt * 16 + i + ((lane >> 4) << 3);
        if (q < NTOK)
          out[((size_t)w * NTOK + q) * CC + h * DH + col] = (_Float16)c[i];
      }
    }
  }
}

// ---------------- launcher ----------------
extern "C" void kernel_launch(void* const* d_in, const int* in_sizes, int n_in,
                              void* d_out, int out_size, void* d_ws, size_t ws_size,
                              hipStream_t stream) {
  (void)in_sizes; (void)n_in; (void)out_size; (void)ws_size;
  const float* x      = (const float*)d_in[0];
  const float* n1w    = (const float*)d_in[1];
  const float* n1b    = (const float*)d_in[2];
  const float* qkv_w  = (const float*)d_in[3];
  const float* qkv_b  = (const float*)d_in[4];
  const float* proj_w = (const float*)d_in[5];
  const float* proj_b = (const float*)d_in[6];
  const float* rpb    = (const float*)d_in[7];
  const float* n2w    = (const float*)d_in[8];
  const float* n2b    = (const float*)d_in[9];
  const float* fc1_w  = (const float*)d_in[10];
  const float* fc1_b  = (const float*)d_in[11];
  const float* fc2_w  = (const float*)d_in[12];
  const float* fc2_b  = (const float*)d_in[13];
  float* outF = (float*)d_out;

  char* ws = (char*)d_ws;
  size_t off = 0;
  auto carve = [&](size_t bytes) { void* p = ws + off; off += (bytes + 255) & ~(size_t)255; return p; };
  _Float16* qkv_wt  = (_Float16*)carve((size_t)QKV_N * CC * 2);
  _Float16* proj_wt = (_Float16*)carve((size_t)CC * CC * 2);
  _Float16* fc1_wt  = (_Float16*)carve((size_t)HID * CC * 2);
  _Float16* fc2_wt  = (_Float16*)carve((size_t)CC * HID * 2);
  _Float16* bufA    = (_Float16*)carve((size_t)TOKENS * CC * 2);   // Xwin -> attn_out -> ln2
  _Float16* bufB    = (_Float16*)carve((size_t)TOKENS * HID * 2);  // qkv -> h1
  float*    x1      = (float*)carve((size_t)TOKENS * CC * 4);

  // 1) convert + transpose weights to f16 [N][K]
  cvt_w_kernel<<<(CC * QKV_N + 255) / 256, 256, 0, stream>>>(qkv_w,  qkv_wt,  CC,  QKV_N);
  cvt_w_kernel<<<(CC * CC    + 255) / 256, 256, 0, stream>>>(proj_w, proj_wt, CC,  CC);
  cvt_w_kernel<<<(CC * HID   + 255) / 256, 256, 0, stream>>>(fc1_w,  fc1_wt,  CC,  HID);
  cvt_w_kernel<<<(HID * CC   + 255) / 256, 256, 0, stream>>>(fc2_w,  fc2_wt,  HID, CC);

  // 2) LN1 + cyclic shift + window partition -> f16
  ln_kernel<true><<<TOKENS, 128, 0, stream>>>(x, n1w, n1b, bufA);

  // 3) QKV GEMM (100352 x 1152 x 384)
  gemm_wmma<EPI_QKV><<<dim3(TOKENS / 128, QKV_N / 128), 256, 0, stream>>>(
      bufA, qkv_wt, qkv_b, nullptr, bufB, QKV_N, CC);

  // 4) windowed attention (2048 windows x 12 heads; 2 heads per block)
  attn_kernel<<<dim3(BATCH * NWIN, HEADS / 2), 64, 0, stream>>>(bufB, rpb, bufA);

  // 5) proj GEMM + window reverse + un-shift + residual -> x1 (f32, spatial layout)
  gemm_wmma<EPI_PROJ><<<dim3(TOKENS / 128, CC / 128), 256, 0, stream>>>(
      bufA, proj_wt, proj_b, x, x1, CC, CC);

  // 6) LN2 -> f16 (spatial order)
  ln_kernel<false><<<TOKENS, 128, 0, stream>>>(x1, n2w, n2b, bufA);

  // 7) fc1 GEMM + exact GELU -> f16
  gemm_wmma<EPI_FC1><<<dim3(TOKENS / 128, HID / 128), 256, 0, stream>>>(
      bufA, fc1_wt, fc1_b, nullptr, bufB, HID, CC);

  // 8) fc2 GEMM + residual(x1) -> d_out (f32)
  gemm_wmma<EPI_FC2><<<dim3(TOKENS / 128, CC / 128), 256, 0, stream>>>(
      bufB, fc2_wt, fc2_b, x1, outF, CC, HID);
}
